// SingleStreamBlock_2731599200663
// MI455X (gfx1250) — compile-verified, matrix operand order checked by reading, not run on hardware
//
#include <hip/hip_runtime.h>
#include <hip/hip_bf16.h>

// ---------------------------------------------------------------------------
// SingleStreamBlock for MI455X (gfx1250, wave32, WMMA bf16 16x16x32)
// B=1, L=2048, D=3072, NH=24, HD=128, MLP=12288
// ---------------------------------------------------------------------------

typedef __attribute__((ext_vector_type(16))) __bf16 v16bf;
typedef __attribute__((ext_vector_type(8)))  __bf16 v8bf;
typedef __attribute__((ext_vector_type(8)))  float  v8f;
typedef __attribute__((ext_vector_type(4)))  int    v4i;

#define LQ    2048
#define DIM   3072
#define NHEAD 24
#define HDIM  128
#define MLPD  12288
#define N1    21504   // 3*DIM + MLPD   (gemm1 N)
#define QKVN  9216    // 3*DIM
#define CATN  15360   // DIM + MLPD     (gemm2 K)

// ---- CDNA5 async global->LDS path (ASYNCcnt), with safe fallback ----------
#if defined(__gfx1250__) && \
    __has_builtin(__builtin_amdgcn_global_load_async_to_lds_b128) && \
    __has_builtin(__builtin_amdgcn_s_wait_asynccnt)
#define USE_ASYNC_LDS 1
#else
#define USE_ASYNC_LDS 0
#endif

#if USE_ASYNC_LDS
__device__ __forceinline__ void async_copy16(const __bf16* g, __bf16* l) {
    __builtin_amdgcn_global_load_async_to_lds_b128(
        (__attribute__((address_space(1))) v4i*)(g),
        (__attribute__((address_space(3))) v4i*)(l),
        /*offset=*/0, /*cpol=*/0);
}
__device__ __forceinline__ void async_wait0() {
    __builtin_amdgcn_s_wait_asynccnt(0);
}
#else
__device__ __forceinline__ void async_copy16(const __bf16* g, __bf16* l) {
    *(v8bf*)l = *(const v8bf*)g;
}
__device__ __forceinline__ void async_wait0() {}
#endif

__device__ __forceinline__ v8f wmma_bf16(v16bf a, v16bf b, v8f c) {
    // D = A(16x32) * B(32x16) + C, f32 accumulate
    return __builtin_amdgcn_wmma_f32_16x16x32_bf16(
        /*neg_a=*/false, a, /*neg_b=*/false, b,
        /*c_mod=*/(short)0, c, /*reuse_a=*/false, /*reuse_b=*/false);
}

union Frag { v16bf v; v8bf h[2]; };
struct Acc8 { v8f c[4][2]; };

// ---------------------------------------------------------------------------
// 128x128 output tile GEMM mainloop.  A: MxK row-major (lda=K),
// B: NxK row-major (ldb=K)  ("NT" gemm; B pre-transposed at convert time).
// 256 threads = 8 waves; wave grid 2(M) x 4(N); per-wave 64x32 = 4x2 WMMA tiles.
// Tile fills use GLOBAL_LOAD_ASYNC_TO_LDS_B128 (ASYNCcnt) on gfx1250.
// ---------------------------------------------------------------------------
__device__ __forceinline__ void gemm_tile_128x128(
    const __bf16* __restrict__ A, const __bf16* __restrict__ B,
    int K, int m0, int n0, Acc8& acc)
{
    constexpr int TK  = 64;
    constexpr int LDT = TK + 8;          // padded LDS stride (bf16 elems)
    __shared__ __bf16 sA[128 * LDT];
    __shared__ __bf16 sB[128 * LDT];

    const int tid  = threadIdx.x;
    const int wave = tid >> 5;
    const int lane = tid & 31;
    const int lmod = lane & 15;
    const int lh   = lane >> 4;
    const int wr   = wave & 1;           // M wave coord (x64)
    const int wc   = wave >> 1;          // N wave coord (x32)

    const v8f zero = {0,0,0,0,0,0,0,0};
#pragma unroll
    for (int i = 0; i < 4; ++i)
#pragma unroll
        for (int j = 0; j < 2; ++j) acc.c[i][j] = zero;

    for (int k0 = 0; k0 < K; k0 += TK) {
        __syncthreads();
        // 128x64 bf16 tile = 1024 16B chunks; 4 per thread, coalesced.
#pragma unroll
        for (int i = 0; i < 4; ++i) {
            int chunk = tid + i * 256;
            int r  = chunk >> 3;
            int cc = (chunk & 7) * 8;
            async_copy16(&A[(size_t)(m0 + r) * K + k0 + cc], &sA[r * LDT + cc]);
            async_copy16(&B[(size_t)(n0 + r) * K + k0 + cc], &sB[r * LDT + cc]);
        }
        if (k0 + TK < K) {               // global_prefetch_b8 of next tiles
            __builtin_prefetch(&A[(size_t)(m0 + (tid >> 3)) * K + k0 + TK + (tid & 7) * 8], 0, 1);
            __builtin_prefetch(&B[(size_t)(n0 + (tid >> 3)) * K + k0 + TK + (tid & 7) * 8], 0, 1);
        }
        async_wait0();                   // drain this wave's ASYNCcnt
        __syncthreads();                 // all waves' tiles now visible

#pragma unroll
        for (int kk = 0; kk < 2; ++kk) {         // two K=32 WMMA steps
            Frag a[4], b[2];
#pragma unroll
            for (int fm = 0; fm < 4; ++fm) {
                int row = 64 * wr + 16 * fm + lmod;
                const __bf16* p = &sA[row * LDT + 32 * kk + 8 * lh];
                a[fm].h[0] = *(const v8bf*)p;
                a[fm].h[1] = *(const v8bf*)(p + 16);
            }
#pragma unroll
            for (int fn = 0; fn < 2; ++fn) {
                int row = 32 * wc + 16 * fn + lmod;
                const __bf16* p = &sB[row * LDT + 32 * kk + 8 * lh];
                b[fn].h[0] = *(const v8bf*)p;
                b[fn].h[1] = *(const v8bf*)(p + 16);
            }
#pragma unroll
            for (int fm = 0; fm < 4; ++fm)
#pragma unroll
                for (int fn = 0; fn < 2; ++fn)
                    acc.c[fm][fn] = wmma_bf16(a[fm].v, b[fn].v, acc.c[fm][fn]);
        }
    }
}

// ---------------------------------------------------------------------------
// fp32 -> bf16 transpose-convert:  in (R x C, f32)  ->  out (C x R, bf16)
// ---------------------------------------------------------------------------
__global__ __launch_bounds__(256) void transpose_bf16_kernel(
    const float* __restrict__ in, __bf16* __restrict__ out, int R, int C)
{
    __shared__ float t[32][33];
    int tx = threadIdx.x & 31, ty = threadIdx.x >> 5;    // ty 0..7
    int r0 = blockIdx.y * 32, c0 = blockIdx.x * 32;
#pragma unroll
    for (int i = 0; i < 4; ++i)
        t[ty + 8 * i][tx] = in[(size_t)(r0 + ty + 8 * i) * C + c0 + tx];
    __syncthreads();
#pragma unroll
    for (int i = 0; i < 4; ++i)
        out[(size_t)(c0 + ty + 8 * i) * R + r0 + tx] = (__bf16)t[tx][ty + 8 * i];
}

// ---------------------------------------------------------------------------
// mod = silu(vec) @ mod_w + mod_b     (1x3072 @ 3072x9216)
// ---------------------------------------------------------------------------
__global__ __launch_bounds__(256) void mod_kernel(
    const float* __restrict__ vec, const float* __restrict__ mod_w,
    const float* __restrict__ mod_b, float* __restrict__ modo)
{
    __shared__ float sv[DIM];
    int tid = threadIdx.x;
    for (int i = tid; i < DIM; i += 256) {
        float v = vec[i];
        sv[i] = v / (1.0f + __expf(-v));
    }
    __syncthreads();
    int col = blockIdx.x * 256 + tid;
    float acc = 0.f;
    for (int i = 0; i < DIM; ++i)
        acc = fmaf(sv[i], mod_w[(size_t)i * QKVN + col], acc);
    modo[col] = acc + mod_b[col];
}

// ---------------------------------------------------------------------------
// x_mod = shift + (1+scale) * layernorm(x)   -> bf16   (one block per row)
// ---------------------------------------------------------------------------
__global__ __launch_bounds__(256) void lnmod_kernel(
    const float* __restrict__ x, const float* __restrict__ mod,
    __bf16* __restrict__ xmod)
{
    int l = blockIdx.x, tid = threadIdx.x;
    const float* xr = x + (size_t)l * DIM;
    __shared__ float red[256];
    float v[12];
    float s = 0.f;
#pragma unroll
    for (int i = 0; i < 12; ++i) { v[i] = xr[tid + 256 * i]; s += v[i]; }
    red[tid] = s; __syncthreads();
    for (int st = 128; st > 0; st >>= 1) {
        if (tid < st) red[tid] += red[tid + st];
        __syncthreads();
    }
    float mu = red[0] * (1.0f / DIM);
    __syncthreads();
    float s2 = 0.f;
#pragma unroll
    for (int i = 0; i < 12; ++i) { float d = v[i] - mu; s2 += d * d; }
    red[tid] = s2; __syncthreads();
    for (int st = 128; st > 0; st >>= 1) {
        if (tid < st) red[tid] += red[tid + st];
        __syncthreads();
    }
    float rs = rsqrtf(red[0] * (1.0f / DIM) + 1e-6f);
    const float* shift = mod;
    const float* scale = mod + DIM;
#pragma unroll
    for (int i = 0; i < 12; ++i) {
        int c = tid + 256 * i;
        float xm = shift[c] + (1.0f + scale[c]) * (v[i] - mu) * rs;
        xmod[(size_t)l * DIM + c] = (__bf16)xm;
    }
}

// ---------------------------------------------------------------------------
// GEMM1: h = x_mod @ w1 + b1.  qkv cols -> f32 buffer; mlp cols -> gelu -> cat
// ---------------------------------------------------------------------------
__global__ __launch_bounds__(256) void gemm1_kernel(
    const __bf16* __restrict__ xmod, const __bf16* __restrict__ w1t,
    const float* __restrict__ b1, float* __restrict__ qkv,
    __bf16* __restrict__ cat)
{
    int n0 = blockIdx.x * 128, m0 = blockIdx.y * 128;
    Acc8 acc;
    gemm_tile_128x128(xmod, w1t, DIM, m0, n0, acc);

    const int tid = threadIdx.x, wave = tid >> 5, lane = tid & 31;
    const int lmod = lane & 15, lh = lane >> 4, wr = wave & 1, wc = wave >> 1;
#pragma unroll
    for (int fm = 0; fm < 4; ++fm)
#pragma unroll
        for (int fn = 0; fn < 2; ++fn)
#pragma unroll
            for (int j = 0; j < 8; ++j) {
                int gm = m0 + 64 * wr + 16 * fm + 8 * lh + j;
                int gn = n0 + 32 * wc + 16 * fn + lmod;
                float v = acc.c[fm][fn][j] + b1[gn];
                if (gn < QKVN) {
                    qkv[(size_t)gm * QKVN + gn] = v;
                } else {
                    float g = 0.5f * v *
                        (1.0f + tanhf(0.7978845608028654f * (v + 0.044715f * v * v * v)));
                    cat[(size_t)gm * CATN + DIM + (gn - QKVN)] = (__bf16)g;
                }
            }
}

// ---------------------------------------------------------------------------
// RMSNorm + RoPE for q,k; repack q,k as [head][l][d], v as [head][d][l] (bf16)
// ---------------------------------------------------------------------------
__global__ __launch_bounds__(128) void qkvprep_kernel(
    const float* __restrict__ qkv, const float* __restrict__ qsc,
    const float* __restrict__ ksc, const float* __restrict__ pe,
    __bf16* __restrict__ Q, __bf16* __restrict__ K, __bf16* __restrict__ Vt)
{
    int l = blockIdx.x, n = blockIdx.y, d = threadIdx.x;
    size_t base = (size_t)l * QKVN + n * HDIM + d;
    float q = qkv[base];
    float k = qkv[base + DIM];
    float v = qkv[base + 2 * DIM];

    __shared__ float r0[128], r1[128];
    r0[d] = q * q; r1[d] = k * k; __syncthreads();
    for (int s = 64; s > 0; s >>= 1) {
        if (d < s) { r0[d] += r0[d + s]; r1[d] += r1[d + s]; }
        __syncthreads();
    }
    float rq = rsqrtf(r0[0] * (1.0f / HDIM) + 1e-6f);
    float rk = rsqrtf(r1[0] * (1.0f / HDIM) + 1e-6f);
    __syncthreads();

    __shared__ float aq[128], ak[128];
    aq[d] = q * rq * qsc[d];
    ak[d] = k * rk * ksc[d];
    __syncthreads();

    int p = d >> 1, j = d & 1;
    size_t pb = ((((size_t)l * 64 + p) * 2) + j) * 2;    // pe[0,0,l,p,j,:]
    float pe0 = pe[pb], pe1 = pe[pb + 1];
    float qr = pe0 * aq[2 * p] + pe1 * aq[2 * p + 1];
    float kr = pe0 * ak[2 * p] + pe1 * ak[2 * p + 1];

    size_t o = ((size_t)n * LQ + l) * HDIM + d;
    Q[o] = (__bf16)qr;
    K[o] = (__bf16)kr;
    Vt[((size_t)n * HDIM + d) * LQ + l] = (__bf16)v;
}

// ---------------------------------------------------------------------------
// Flash attention (full, non-causal). Block = 128 Q rows x one head, 8 waves.
// Each wave owns 16 Q rows; keys streamed in 128-wide chunks through LDS
// via async-to-LDS loads.
// ---------------------------------------------------------------------------
__global__ __launch_bounds__(256) void attn_kernel(
    const __bf16* __restrict__ Q, const __bf16* __restrict__ K,
    const __bf16* __restrict__ Vt, __bf16* __restrict__ cat)
{
    constexpr int LDT = HDIM + 8;            // 136
    __shared__ __bf16 sK[128 * LDT];
    __shared__ __bf16 sV[128 * LDT];         // [d][key]
    __shared__ __bf16 sP[128 * LDT];         // per-wave 16-row regions

    const int head = blockIdx.y, m0 = blockIdx.x * 128;
    const __bf16* Qh = Q  + (size_t)head * LQ * HDIM;
    const __bf16* Kh = K  + (size_t)head * LQ * HDIM;
    const __bf16* Vh = Vt + (size_t)head * HDIM * LQ;

    const int tid = threadIdx.x, wave = tid >> 5, lane = tid & 31;
    const int lmod = lane & 15, lh = lane >> 4;

    // Q fragments held in registers for the whole kernel (16 rows x 128 d)
    Frag qf[4];
    const int qrow = m0 + 16 * wave + lmod;
#pragma unroll
    for (int ks = 0; ks < 4; ++ks) {
        const __bf16* p = &Qh[(size_t)qrow * HDIM + 32 * ks + 8 * lh];
        qf[ks].h[0] = *(const v8bf*)p;
        qf[ks].h[1] = *(const v8bf*)(p + 16);
    }

    const v8f zero = {0,0,0,0,0,0,0,0};
    float mrow[8], lrow[8];
    v8f o[8];
#pragma unroll
    for (int j = 0; j < 8; ++j) { mrow[j] = -1e30f; lrow[j] = 0.f; }
#pragma unroll
    for (int dt = 0; dt < 8; ++dt) o[dt] = zero;

    const float sc = 0.08838834764831845f;   // 1/sqrt(128)

    for (int kb = 0; kb < LQ; kb += 128) {
        __syncthreads();
#pragma unroll
        for (int i = 0; i < 8; ++i) {        // 2x 32KB tiles, 8 chunks/thread
            int chunk = tid + i * 256;
            int r = chunk >> 4;
            int c = (chunk & 15) * 8;
            async_copy16(&Kh[(size_t)(kb + r) * HDIM + c], &sK[r * LDT + c]);
            async_copy16(&Vh[(size_t)r * LQ + kb + c],     &sV[r * LDT + c]);
        }
        async_wait0();
        __syncthreads();

        // S = Q @ K^T for this wave's 16 rows x 128 keys
        v8f s[8];
#pragma unroll
        for (int nt = 0; nt < 8; ++nt) {
            s[nt] = zero;
#pragma unroll
            for (int ks = 0; ks < 4; ++ks) {
                Frag bf;
                const __bf16* p = &sK[(16 * nt + lmod) * LDT + 32 * ks + 8 * lh];
                bf.h[0] = *(const v8bf*)p;
                bf.h[1] = *(const v8bf*)(p + 16);
                s[nt] = wmma_bf16(qf[ks].v, bf.v, s[nt]);
            }
        }

        // online softmax (row stats live in 16-lane halves, slot j = row)
#pragma unroll
        for (int j = 0; j < 8; ++j) {
            float mx = -1e30f;
#pragma unroll
            for (int nt = 0; nt < 8; ++nt) mx = fmaxf(mx, s[nt][j]);
#pragma unroll
            for (int off = 1; off < 16; off <<= 1)
                mx = fmaxf(mx, __shfl_xor(mx, off, 16));
            float mnew  = fmaxf(mrow[j], mx * sc);
            float alpha = __expf(mrow[j] - mnew);
            float rs = 0.f;
#pragma unroll
            for (int nt = 0; nt < 8; ++nt) {
                float pv = __expf(sc * s[nt][j] - mnew);
                rs += pv;
                sP[(16 * wave + j + 8 * lh) * LDT + 16 * nt + lmod] = (__bf16)pv;
            }
#pragma unroll
            for (int off = 1; off < 16; off <<= 1)
                rs += __shfl_xor(rs, off, 16);
            lrow[j] = lrow[j] * alpha + rs;
            mrow[j] = mnew;
#pragma unroll
            for (int dt = 0; dt < 8; ++dt) o[dt][j] = o[dt][j] * alpha;
        }

        // O += P @ V   (P re-read from LDS in A-fragment layout; same wave)
#pragma unroll
        for (int ks = 0; ks < 4; ++ks) {
            Frag pf;
            const __bf16* p = &sP[(16 * wave + lmod) * LDT + 32 * ks + 8 * lh];
            pf.h[0] = *(const v8bf*)p;
            pf.h[1] = *(const v8bf*)(p + 16);
#pragma unroll
            for (int dt = 0; dt < 8; ++dt) {
                Frag vf;
                const __bf16* pv2 = &sV[(16 * dt + lmod) * LDT + 32 * ks + 8 * lh];
                vf.h[0] = *(const v8bf*)pv2;
                vf.h[1] = *(const v8bf*)(pv2 + 16);
                o[dt] = wmma_bf16(pf.v, vf.v, o[dt]);
            }
        }
    }

    // finalize: divide by row sums, write bf16 into concat buffer cols [0, D)
#pragma unroll
    for (int j = 0; j < 8; ++j) {
        float inv = 1.0f / lrow[j];
        int row = m0 + 16 * wave + j + 8 * lh;
#pragma unroll
        for (int dt = 0; dt < 8; ++dt) {
            int d = 16 * dt + lmod;
            cat[(size_t)row * CATN + head * HDIM + d] = (__bf16)(o[dt][j] * inv);
        }
    }
}

// ---------------------------------------------------------------------------
// GEMM2: out = x + gate * (cat @ w2 + b2)
// ---------------------------------------------------------------------------
__global__ __launch_bounds__(256) void gemm2_kernel(
    const __bf16* __restrict__ cat, const __bf16* __restrict__ w2t,
    const float* __restrict__ b2, const float* __restrict__ x,
    const float* __restrict__ mod, float* __restrict__ out)
{
    int n0 = blockIdx.x * 128, m0 = blockIdx.y * 128;
    Acc8 acc;
    gemm_tile_128x128(cat, w2t, CATN, m0, n0, acc);

    const float* gate = mod + 2 * DIM;
    const int tid = threadIdx.x, wave = tid >> 5, lane = tid & 31;
    const int lmod = lane & 15, lh = lane >> 4, wr = wave & 1, wc = wave >> 1;
#pragma unroll
    for (int fm = 0; fm < 4; ++fm)
#pragma unroll
        for (int fn = 0; fn < 2; ++fn)
#pragma unroll
            for (int j = 0; j < 8; ++j) {
                int gm = m0 + 64 * wr + 16 * fm + 8 * lh + j;
                int gn = n0 + 32 * wc + 16 * fn + lmod;
                float v = acc.c[fm][fn][j] + b2[gn];
                size_t idx = (size_t)gm * DIM + gn;
                out[idx] = x[idx] + gate[gn] * v;
            }
}

// ---------------------------------------------------------------------------
// Host launcher
// ---------------------------------------------------------------------------
extern "C" void kernel_launch(void* const* d_in, const int* in_sizes, int n_in,
                              void* d_out, int out_size, void* d_ws, size_t ws_size,
                              hipStream_t stream)
{
    const float* x     = (const float*)d_in[1];
    const float* vec   = (const float*)d_in[2];
    const float* pe    = (const float*)d_in[3];
    const float* mod_w = (const float*)d_in[4];
    const float* mod_b = (const float*)d_in[5];
    const float* w1    = (const float*)d_in[6];
    const float* b1    = (const float*)d_in[7];
    const float* w2    = (const float*)d_in[8];
    const float* b2    = (const float*)d_in[9];
    const float* qsc   = (const float*)d_in[10];
    const float* ksc   = (const float*)d_in[11];
    float* out = (float*)d_out;

    char* p = (char*)d_ws;
    auto alloc = [&](size_t bytes) -> char* {
        char* r = p;
        p += (bytes + 255) & ~(size_t)255;
        return r;
    };
    float*  modbuf = (float*) alloc((size_t)QKVN * sizeof(float));
    __bf16* xmod   = (__bf16*)alloc((size_t)LQ * DIM * 2);
    __bf16* w1t    = (__bf16*)alloc((size_t)N1 * DIM * 2);
    __bf16* w2t    = (__bf16*)alloc((size_t)DIM * CATN * 2);
    float*  qkv    = (float*) alloc((size_t)LQ * QKVN * 4);
    __bf16* Qb     = (__bf16*)alloc((size_t)NHEAD * LQ * HDIM * 2);
    __bf16* Kb     = (__bf16*)alloc((size_t)NHEAD * LQ * HDIM * 2);
    __bf16* Vtb    = (__bf16*)alloc((size_t)NHEAD * LQ * HDIM * 2);
    __bf16* cat    = (__bf16*)alloc((size_t)LQ * CATN * 2);

    // weights fp32 -> bf16, transposed so both GEMM operands are K-contiguous
    transpose_bf16_kernel<<<dim3(N1 / 32, DIM / 32), 256, 0, stream>>>(w1, w1t, DIM, N1);
    transpose_bf16_kernel<<<dim3(DIM / 32, CATN / 32), 256, 0, stream>>>(w2, w2t, CATN, DIM);

    mod_kernel<<<QKVN / 256, 256, 0, stream>>>(vec, mod_w, mod_b, modbuf);
    lnmod_kernel<<<LQ, 256, 0, stream>>>(x, modbuf, xmod);
    gemm1_kernel<<<dim3(N1 / 128, LQ / 128), 256, 0, stream>>>(xmod, w1t, b1, qkv, cat);
    qkvprep_kernel<<<dim3(LQ, NHEAD), 128, 0, stream>>>(qkv, qsc, ksc, pe, Qb, Kb, Vtb);
    attn_kernel<<<dim3(LQ / 128, NHEAD), 256, 0, stream>>>(Qb, Kb, Vtb, cat);
    gemm2_kernel<<<dim3(DIM / 128, LQ / 128), 256, 0, stream>>>(cat, w2t, b2, x, modbuf, out);
}
